// CROMA_84009560310457
// MI455X (gfx1250) — matrix operand order repaired and checked by reading, not compiled, not run on hardware
//
#include <hip/hip_runtime.h>

// ---------------------------------------------------------------------------
// MI455X (gfx1250) ViT forward pass.
// GEMMs + attention matmuls on V_WMMA_F32_16X16X32_BF16 (wave32).
// LDS tiles staged by the Tensor Data Mover (tensor_load_to_lds, TENSORcnt),
// double-buffered in the GEMM so DMA overlaps WMMA.
// ---------------------------------------------------------------------------

typedef unsigned short u16;
typedef __bf16 bf16x16 __attribute__((ext_vector_type(16)));
typedef float  f32x8   __attribute__((ext_vector_type(8)));
typedef unsigned int u32x4 __attribute__((ext_vector_type(4)));
typedef int i32x8 __attribute__((ext_vector_type(8)));
typedef int i32x4 __attribute__((ext_vector_type(4)));

union Frag { bf16x16 v; u32x4 q[2]; };

#define DEV static __device__ __forceinline__

#define NSEQ   1024
#define DIMM   768
#define FFD    3072
#define MROWS  4096          // B * NSEQ
#define NHEAD  16
#define DHPAD  64            // head dim 48 padded to 64
#define SCALE_ 0.14433756729740643f   // 48^-0.5

#if __has_builtin(__builtin_amdgcn_tensor_load_to_lds)
#define HAVE_TDM 1
#else
#define HAVE_TDM 0
#endif

DEV u16 f2bf(float f) {
  union { float f; unsigned u; } v; v.f = f;
  unsigned r = v.u + 0x7FFFu + ((v.u >> 16) & 1u);   // round-nearest-even
  return (u16)(r >> 16);
}

DEV f32x8 wmma_bf16(const Frag& a, const Frag& b, f32x8 c) {
  return __builtin_amdgcn_wmma_f32_16x16x32_bf16(false, a.v, false, b.v,
                                                 (short)0, c, false, false);
}

DEV unsigned lds_off(const void* p) { return (unsigned)(size_t)p; }

#if HAVE_TDM
// 2D tile DMA: tile0 elems (contiguous, 2B each) x tile1 rows, row stride
// stride0 elems; LDS gets pad_amount((code+1) dwords) after every
// 2^(code+1) dwords -> matches our padded LDS row strides.
DEV void tdm_load_2d(unsigned lds_addr, const void* gaddr,
                     unsigned tile0, unsigned tile1,
                     unsigned dim0, unsigned stride0,
                     unsigned pad_int_code, unsigned pad_amt_code) {
  unsigned long long ga = (unsigned long long)(size_t)gaddr;
  u32x4 g0;
  g0[0] = 1u;                                            // count=1 valid D#
  g0[1] = lds_addr;                                      // LDS byte address
  g0[2] = (unsigned)(ga & 0xFFFFFFFFu);                  // global addr lo
  g0[3] = (unsigned)((ga >> 32) & 0x01FFFFFFu) | 0x80000000u;  // hi + type=2
  i32x8 g1;
  g1[0] = (int)((1u << 16) |                             // data_size = 2B
                (1u << 20) |                             // pad_enable
                (pad_int_code << 22) | (pad_amt_code << 25));
  g1[1] = (int)((dim0 & 0xFFFFu) << 16);                 // tensor_dim0 lo16
  g1[2] = (int)((dim0 >> 16) | ((tile1 & 0xFFFFu) << 16)); // dim0 hi | dim1 lo
  g1[3] = (int)((tile1 >> 16) | (tile0 << 16));          // dim1 hi | tile_dim0
  g1[4] = (int)(tile1 & 0xFFFFu);                        // tile_dim1 (tile2=0)
  g1[5] = (int)stride0;                                  // dim0 stride lo32
  g1[6] = 0;
  g1[7] = 0;
  i32x4 z4 = {0, 0, 0, 0};
  i32x8 z8v = {0, 0, 0, 0, 0, 0, 0, 0};
  __builtin_amdgcn_tensor_load_to_lds(g0, g1, z4, z4, z8v, 0);
}
#endif

// ---------------------------------------------------------------------------
// GEMM: C[M,N] (f32, += if RES) = A[M,K](bf16) * Bt[N,K](bf16)^T + bias
// block tile 128x128x64; 8 waves, wave tile 32x64; TDM double-buffered LDS.
// ---------------------------------------------------------------------------
template <bool RES>
__global__ __launch_bounds__(256) void gemm_bf16(const u16* __restrict__ A,
                                                 const u16* __restrict__ Bt,
                                                 const float* __restrict__ bias,
                                                 float* __restrict__ C,
                                                 int N, int K) {
  __shared__ __align__(16) u16 sA[2][128 * 72];   // 64 K + 8 pad per row
  __shared__ __align__(16) u16 sB[2][128 * 72];
  const int tid  = threadIdx.x;
  const int wave = tid >> 5, lane = tid & 31;
  const int rrow = lane & 15, half = lane >> 4;
  const int aof0 = half * 8, aof1 = 16 + half * 8;       // A-operand K chunks
  const int bof0 = half * 8, bof1 = 16 + half * 8;       // B-operand K chunks
  const int m0 = blockIdx.y * 128, n0 = blockIdx.x * 128;
  const int mB = (wave >> 1) * 32, nB = (wave & 1) * 64;
  const int nk = K >> 6;

  const f32x8 z8 = {0.f, 0.f, 0.f, 0.f, 0.f, 0.f, 0.f, 0.f};
  f32x8 acc[2][4];
#pragma unroll
  for (int i = 0; i < 2; ++i)
#pragma unroll
    for (int j = 0; j < 4; ++j) acc[i][j] = z8;

#if HAVE_TDM
  if (tid == 0) {   // wave 0 primes tile 0 (TDM ignores EXEC)
    tdm_load_2d(lds_off(&sA[0][0]), A + (size_t)m0 * K, 64, 128, (unsigned)K, (unsigned)K, 4, 3);
    tdm_load_2d(lds_off(&sB[0][0]), Bt + (size_t)n0 * K, 64, 128, (unsigned)K, (unsigned)K, 4, 3);
  }
#endif

  for (int kt = 0; kt < nk; ++kt) {
    const int cur = kt & 1;
#if HAVE_TDM
    if (tid == 0) {
      if (kt + 1 < nk) {   // overlap next tile's DMA with this tile's WMMAs
        tdm_load_2d(lds_off(&sA[cur ^ 1][0]), A + (size_t)m0 * K + (kt + 1) * 64,
                    64, 128, (unsigned)K, (unsigned)K, 4, 3);
        tdm_load_2d(lds_off(&sB[cur ^ 1][0]), Bt + (size_t)n0 * K + (kt + 1) * 64,
                    64, 128, (unsigned)K, (unsigned)K, 4, 3);
        __builtin_amdgcn_s_wait_tensorcnt((short)2);   // in-order: tile kt done
      } else {
        __builtin_amdgcn_s_wait_tensorcnt((short)0);
      }
    }
    __syncthreads();
#else
    __syncthreads();
    const int k0 = kt * 64;
#pragma unroll
    for (int i = 0; i < 4; ++i) {        // 1024 16B chunks each, 4/thread
      int id = tid + i * 256;
      int r = id >> 3, ch = (id & 7) * 8;
      *(u32x4*)&sA[cur][r * 72 + ch] = *(const u32x4*)(A + (size_t)(m0 + r) * K + k0 + ch);
      *(u32x4*)&sB[cur][r * 72 + ch] = *(const u32x4*)(Bt + (size_t)(n0 + r) * K + k0 + ch);
    }
    __syncthreads();
#endif

#pragma unroll
    for (int ks = 0; ks < 2; ++ks) {     // two 16x16x32 K-steps per stage
      const int kofs = ks * 32;
      Frag aF[2];
#pragma unroll
      for (int mi = 0; mi < 2; ++mi) {
        const u16* rp = &sA[cur][(mB + mi * 16 + rrow) * 72 + kofs];
        aF[mi].q[0] = *(const u32x4*)(rp + aof0);
        aF[mi].q[1] = *(const u32x4*)(rp + aof1);
      }
#pragma unroll
      for (int ni = 0; ni < 4; ++ni) {
        Frag bF;
        const u16* rp = &sB[cur][(nB + ni * 16 + rrow) * 72 + kofs];
        bF.q[0] = *(const u32x4*)(rp + bof0);
        bF.q[1] = *(const u32x4*)(rp + bof1);
        acc[0][ni] = wmma_bf16(aF[0], bF, acc[0][ni]);
        acc[1][ni] = wmma_bf16(aF[1], bF, acc[1][ni]);
      }
    }
    __syncthreads();
  }

  // epilogue: C layout — vgpr r: row = r + 8*half, col = lane&15
#pragma unroll
  for (int mi = 0; mi < 2; ++mi)
#pragma unroll
    for (int ni = 0; ni < 4; ++ni) {
      int col = n0 + nB + ni * 16 + rrow;
      float bv = bias ? bias[col] : 0.f;
#pragma unroll
      for (int r = 0; r < 8; ++r) {
        int row = m0 + mB + mi * 16 + r + half * 8;
        size_t idx = (size_t)row * N + col;
        float v = acc[mi][ni][r] + bv;
        C[idx] = RES ? (C[idx] + v) : v;
      }
    }
}

// ---------------------------------------------------------------------------
// Flash-style attention. grid = (NSEQ/128, NHEAD, B); 8 waves x 16 q-rows.
// Qh/Kh: [B,H,N,64] bf16 (d padded w/ zeros). Vt: [B,H,64,N] bf16.
// K / Vt blocks staged by TDM; S and P tiles per-wave in LDS.
// ---------------------------------------------------------------------------
__global__ __launch_bounds__(256) void attn_kernel(const u16* __restrict__ Qh,
                                                   const u16* __restrict__ Kh,
                                                   const u16* __restrict__ Vt,
                                                   const float* __restrict__ Bias,
                                                   u16* __restrict__ Out) {
  __shared__ __align__(16) u16 sK[128 * 72];       // 128 keys x 64 d (+pad)
  __shared__ __align__(16) u16 sVt[64 * 136];      // 64 d x 128 keys (+pad)
  __shared__ float sS[8][16 * 128];                // per-wave score tile
  __shared__ __align__(16) u16 sP[8][16 * 136];    // per-wave P (bf16)
  __shared__ float sM[8][16], sL[8][16], sC[8][16];

  const int tid  = threadIdx.x;
  const int wave = tid >> 5, lane = tid & 31;
  const int rrow = lane & 15, half = lane >> 4;
  const int aof0 = half * 8, aof1 = 16 + half * 8;
  const int bof0 = half * 8, bof1 = 16 + half * 8;
  const int h = blockIdx.y, b = blockIdx.z;
  const int q0 = blockIdx.x * 128 + wave * 16;
  const size_t bh = (size_t)(b * NHEAD + h);
  const u16* qb = Qh + bh * NSEQ * DHPAD;
  const u16* kb = Kh + bh * NSEQ * DHPAD;
  const u16* vb = Vt + bh * DHPAD * NSEQ;
  const float* bb = Bias + (size_t)h * NSEQ * NSEQ;

  // Q fragments (A-operand), resident for whole kernel
  Frag qf[2];
#pragma unroll
  for (int kf = 0; kf < 2; ++kf) {
    const u16* rp = qb + (size_t)(q0 + rrow) * DHPAD + kf * 32;
    qf[kf].q[0] = *(const u32x4*)(rp + aof0);
    qf[kf].q[1] = *(const u32x4*)(rp + aof1);
  }

  const f32x8 z8 = {0.f, 0.f, 0.f, 0.f, 0.f, 0.f, 0.f, 0.f};
  f32x8 accO[4];
#pragma unroll
  for (int i = 0; i < 4; ++i) accO[i] = z8;
  if (lane < 16) { sM[wave][lane] = -3.0e38f; sL[wave][lane] = 0.f; }

  for (int kb0 = 0; kb0 < NSEQ; kb0 += 128) {
    __syncthreads();
#if HAVE_TDM
    if (tid == 0) {
      tdm_load_2d(lds_off(&sK[0]), kb + (size_t)kb0 * DHPAD, 64, 128, 64, 64, 4, 3);
      tdm_load_2d(lds_off(&sVt[0]), vb + kb0, 128, 64, 128, NSEQ, 5, 3);
      __builtin_amdgcn_s_wait_tensorcnt((short)0);
    }
#else
#pragma unroll
    for (int i = 0; i < 4; ++i) {        // stage K (1024 chunks) + Vt (1024)
      int id = tid + i * 256;
      int kr = id >> 3, kch = (id & 7) * 8;
      *(u32x4*)&sK[kr * 72 + kch] = *(const u32x4*)(kb + (size_t)(kb0 + kr) * DHPAD + kch);
      int vr = id >> 4, vch = (id & 15) * 8;
      *(u32x4*)&sVt[vr * 136 + vch] = *(const u32x4*)(vb + (size_t)vr * NSEQ + kb0 + vch);
    }
#endif
    __syncthreads();

    // S = Q K^T  (16 x 128 tile per wave)
#pragma unroll
    for (int ni = 0; ni < 8; ++ni) {
      f32x8 s = z8;
#pragma unroll
      for (int kf = 0; kf < 2; ++kf) {
        Frag kF;
        const u16* rp = &sK[(ni * 16 + rrow) * 72 + kf * 32];
        kF.q[0] = *(const u32x4*)(rp + bof0);
        kF.q[1] = *(const u32x4*)(rp + bof1);
        s = wmma_bf16(qf[kf], kF, s);
      }
#pragma unroll
      for (int r = 0; r < 8; ++r)
        sS[wave][(r + half * 8) * 128 + ni * 16 + rrow] = s[r];
    }

    // online softmax: lane owns row rrow, column half [half*64, half*64+64)
    {
      float* Srow = &sS[wave][rrow * 128];
      const float* brow = bb + (size_t)(q0 + rrow) * NSEQ + kb0;
      const int c0 = half * 64;
      float pm = -3.0e38f;
      for (int c = 0; c < 64; ++c) {
        float v = Srow[c0 + c] * SCALE_ + brow[c0 + c];
        Srow[c0 + c] = v;
        pm = fmaxf(pm, v);
      }
      pm = fmaxf(pm, __shfl_xor(pm, 16));
      float mo = sM[wave][rrow];
      float mn = fmaxf(mo, pm);
      float corr = __expf(mo - mn);
      float ps = 0.f;
      u16* Prow = &sP[wave][rrow * 136];
      for (int c = 0; c < 64; ++c) {
        float pv = __expf(Srow[c0 + c] - mn);
        Prow[c0 + c] = f2bf(pv);
        ps += pv;
      }
      ps += __shfl_xor(ps, 16);
      if (!half) {
        sL[wave][rrow] = sL[wave][rrow] * corr + ps;
        sM[wave][rrow] = mn;
        sC[wave][rrow] = corr;
      }
    }

    // rescale running O accumulators
    float cf[8];
#pragma unroll
    for (int r = 0; r < 8; ++r) cf[r] = sC[wave][r + half * 8];
#pragma unroll
    for (int nd = 0; nd < 4; ++nd)
#pragma unroll
      for (int r = 0; r < 8; ++r) accO[nd][r] *= cf[r];

    // O += P @ V
#pragma unroll
    for (int kf = 0; kf < 4; ++kf) {
      Frag pF;
      const u16* rp = &sP[wave][rrow * 136 + kf * 32];
      pF.q[0] = *(const u32x4*)(rp + aof0);
      pF.q[1] = *(const u32x4*)(rp + aof1);
#pragma unroll
      for (int nd = 0; nd < 4; ++nd) {
        Frag vF;
        const u16* vp = &sVt[(nd * 16 + rrow) * 136 + kf * 32];
        vF.q[0] = *(const u32x4*)(vp + bof0);
        vF.q[1] = *(const u32x4*)(vp + bof1);
        accO[nd] = wmma_bf16(pF, vF, accO[nd]);
      }
    }
  }

  // write out (cols 0..47 only; 48..63 are pad)
  float li[8];
#pragma unroll
  for (int r = 0; r < 8; ++r) li[r] = 1.f / sL[wave][r + half * 8];
#pragma unroll
  for (int nd = 0; nd < 3; ++nd)
#pragma unroll
    for (int r = 0; r < 8; ++r) {
      int row = q0 + r + half * 8;
      int col = h * 48 + nd * 16 + rrow;
      Out[(size_t)(b * NSEQ + row) * DIMM + col] = f2bf(accO[nd][r] * li[r]);
    }
}

// ---------------------------------------------------------------------------
// LayerNorm over 768: one block per row, 3 elems/thread.
// ---------------------------------------------------------------------------
template <bool BF16OUT>
__global__ __launch_bounds__(256) void ln_kernel(const float* __restrict__ x,
                                                 const float* __restrict__ g,
                                                 const float* __restrict__ bt,
                                                 void* __restrict__ outp) {
  __shared__ float red[8];
  __shared__ float bc;
  const int row = blockIdx.x, tid = threadIdx.x;
  const float* xr = x + (size_t)row * DIMM;
  float v0 = xr[tid], v1 = xr[tid + 256], v2 = xr[tid + 512];
  float s = v0 + v1 + v2;
#pragma unroll
  for (int o = 16; o > 0; o >>= 1) s += __shfl_xor(s, o);
  if ((tid & 31) == 0) red[tid >> 5] = s;
  __syncthreads();
  if (tid == 0) { float t = 0; for (int w = 0; w < 8; ++w) t += red[w]; bc = t * (1.f / 768.f); }
  __syncthreads();
  float mu = bc;
  float d0 = v0 - mu, d1 = v1 - mu, d2 = v2 - mu;
  float q = d0 * d0 + d1 * d1 + d2 * d2;
#pragma unroll
  for (int o = 16; o > 0; o >>= 1) q += __shfl_xor(q, o);
  if ((tid & 31) == 0) red[tid >> 5] = q;
  __syncthreads();
  if (tid == 0) { float t = 0; for (int w = 0; w < 8; ++w) t += red[w]; bc = rsqrtf(t * (1.f / 768.f) + 1e-5f); }
  __syncthreads();
  float rs = bc;
  float vv[3] = {d0, d1, d2};
#pragma unroll
  for (int i = 0; i < 3; ++i) {
    int idx = tid + i * 256;
    float y = vv[i] * rs * g[idx] + bt[idx];
    if (BF16OUT) ((u16*)outp)[(size_t)row * DIMM + idx] = f2bf(y);
    else         ((float*)outp)[(size_t)row * DIMM + idx] = y;
  }
}

// ---------------------------------------------------------------------------
// small elementwise kernels
// ---------------------------------------------------------------------------
__global__ void convT_kernel(const float* __restrict__ w, u16* __restrict__ wt,
                             int K, int N) {        // w[K][N] -> wt[N][K] bf16
  size_t id = (size_t)blockIdx.x * 256 + threadIdx.x;
  if (id >= (size_t)K * N) return;
  int k = (int)(id / N), n = (int)(id - (size_t)k * N);
  wt[(size_t)n * K + k] = f2bf(w[id]);
}

__global__ void patchify_kernel(const float* __restrict__ imgs, u16* __restrict__ out) {
  int id = blockIdx.x * 256 + threadIdx.x;          // < 4096*768
  int pidx = id / DIMM, col = id - pidx * DIMM;
  int b = pidx >> 10, pi = pidx & 1023, hb = pi >> 5, wb = pi & 31;
  int c = col >> 6, rem = col & 63, ii = rem >> 3, jj = rem & 7;
  size_t src = (((size_t)b * 12 + c) * 256 + hb * 8 + ii) * 256 + wb * 8 + jj;
  out[id] = f2bf(imgs[src]);
}

__global__ void reshape_qkv_kernel(const float* __restrict__ qkv,
                                   u16* __restrict__ qh, u16* __restrict__ kh,
                                   u16* __restrict__ vt) {
  int id = blockIdx.x * 256 + threadIdx.x;          // < 4096*768
  int row = id / DIMM, dim = id - row * DIMM;
  int b = row >> 10, n = row & 1023;
  int h = dim / 48, d = dim - h * 48;
  const float* src = qkv + (size_t)row * (3 * DIMM) + dim;
  size_t bh = (size_t)(b * NHEAD + h);
  size_t base = (bh * NSEQ + n) * DHPAD + d;
  qh[base] = f2bf(src[0]);
  kh[base] = f2bf(src[DIMM]);
  vt[(bh * DHPAD + d) * NSEQ + n] = f2bf(src[2 * DIMM]);
}

__global__ void gelu_kernel(const float* __restrict__ in, u16* __restrict__ out, int n) {
  int id = blockIdx.x * 256 + threadIdx.x;
  if (id < n) {
    float v = in[id];
    out[id] = f2bf(0.5f * v * (1.f + erff(v * 0.70710678118f)));
  }
}

__global__ void fill0_kernel(u16* p, size_t n) {
  size_t id = (size_t)blockIdx.x * 256 + threadIdx.x;
  if (id < n) p[id] = 0;
}

// ---------------------------------------------------------------------------
// host orchestration
// ---------------------------------------------------------------------------
extern "C" void kernel_launch(void* const* d_in, const int* in_sizes, int n_in,
                              void* d_out, int out_size, void* d_ws, size_t ws_size,
                              hipStream_t stream) {
  (void)in_sizes; (void)n_in; (void)out_size; (void)ws_size;
  const float* imgs  = (const float*)d_in[0];
  const float* abias = (const float*)d_in[1];
  const float* w_in  = (const float*)d_in[2];
  const float* b_in  = (const float*)d_in[3];
  const float* ln1_g = (const float*)d_in[4];
  const float* ln1_b = (const float*)d_in[5];
  const float* w_qkv = (const float*)d_in[6];
  const float* w_out = (const float*)d_in[7];
  const float* b_out = (const float*)d_in[8];
  const float* ln2_g = (const float*)d_in[9];
  const float* ln2_b = (const float*)d_in[10];
  const float* w_ff1 = (const float*)d_in[11];
  const float* b_ff1 = (const float*)d_in[12];
  const float* w_ff2 = (const float*)d_in[13];
  const float* b_ff2 = (const float*)d_in[14];
  const float* lnf_g = (const float*)d_in[15];
  const float* lnf_b = (const float*)d_in[16];

  char* wp = (char*)d_ws;
  auto take = [&](size_t bytes) -> void* {
    void* r = (void*)wp;
    wp += (bytes + 255) & ~(size_t)255;
    return r;
  };
  const size_t ST_QKV = (size_t)2304 * 768;   // elements / layer (transposed)
  const size_t ST_OUT = (size_t)768 * 768;
  const size_t ST_FF1 = (size_t)3072 * 768;
  const size_t ST_FF2 = (size_t)768 * 3072;
  const size_t QKV_ELEMS = (size_t)4 * NHEAD * NSEQ * DHPAD;   // per buffer

  u16* wT_in  = (u16*)take((size_t)768 * 768 * 2);
  u16* wT_qkv = (u16*)take(8 * ST_QKV * 2);
  u16* wT_out = (u16*)take(8 * ST_OUT * 2);
  u16* wT_ff1 = (u16*)take(8 * ST_FF1 * 2);
  u16* wT_ff2 = (u16*)take(8 * ST_FF2 * 2);
  float* xbuf = (float*)take((size_t)MROWS * DIMM * 4);
  u16*  abuf  = (u16*)take((size_t)MROWS * FFD * 2);
  float* tbuf = (float*)take((size_t)MROWS * FFD * 4);
  u16* qh = (u16*)take(QKV_ELEMS * 2);       // qh,kh,vt contiguous (sizes %256==0)
  u16* kh = (u16*)take(QKV_ELEMS * 2);
  u16* vt = (u16*)take(QKV_ELEMS * 2);

  auto blocks = [](size_t n) { return (unsigned)((n + 255) / 256); };

  // --- weight conversion (f32 -> bf16, transposed) ---
  convT_kernel<<<blocks((size_t)768 * 768), 256, 0, stream>>>(w_in, wT_in, 768, 768);
  for (int l = 0; l < 8; ++l) {
    convT_kernel<<<blocks((size_t)768 * 2304), 256, 0, stream>>>(
        w_qkv + (size_t)l * 768 * 2304, wT_qkv + l * ST_QKV, 768, 2304);
    convT_kernel<<<blocks((size_t)768 * 768), 256, 0, stream>>>(
        w_out + (size_t)l * 768 * 768, wT_out + l * ST_OUT, 768, 768);
    convT_kernel<<<blocks((size_t)768 * 3072), 256, 0, stream>>>(
        w_ff1 + (size_t)l * 768 * 3072, wT_ff1 + l * ST_FF1, 768, 3072);
    convT_kernel<<<blocks((size_t)3072 * 768), 256, 0, stream>>>(
        w_ff2 + (size_t)l * 3072 * 768, wT_ff2 + l * ST_FF2, 3072, 768);
  }

  // zero q/k/vt (covers d=48..63 pad lanes, never overwritten by reshape)
  fill0_kernel<<<blocks(3 * QKV_ELEMS), 256, 0, stream>>>(qh, 3 * QKV_ELEMS);

  // --- patch embed ---
  patchify_kernel<<<blocks((size_t)MROWS * DIMM), 256, 0, stream>>>(imgs, abuf);
  gemm_bf16<false><<<dim3(DIMM / 128, MROWS / 128), 256, 0, stream>>>(
      abuf, wT_in, b_in, xbuf, DIMM, DIMM);

  // --- transformer layers ---
  for (int l = 0; l < 8; ++l) {
    ln_kernel<true><<<MROWS, 256, 0, stream>>>(xbuf, ln1_g + l * DIMM, ln1_b + l * DIMM, abuf);
    gemm_bf16<false><<<dim3(2304 / 128, MROWS / 128), 256, 0, stream>>>(
        abuf, wT_qkv + l * ST_QKV, nullptr, tbuf, 2304, DIMM);
    reshape_qkv_kernel<<<blocks((size_t)MROWS * DIMM), 256, 0, stream>>>(tbuf, qh, kh, vt);
    attn_kernel<<<dim3(NSEQ / 128, NHEAD, 4), 256, 0, stream>>>(qh, kh, vt, abias, abuf);
    gemm_bf16<true><<<dim3(DIMM / 128, MROWS / 128), 256, 0, stream>>>(
        abuf, wT_out + l * ST_OUT, b_out + l * DIMM, xbuf, DIMM, DIMM);
    ln_kernel<true><<<MROWS, 256, 0, stream>>>(xbuf, ln2_g + l * DIMM, ln2_b + l * DIMM, abuf);
    gemm_bf16<false><<<dim3(FFD / 128, MROWS / 128), 256, 0, stream>>>(
        abuf, wT_ff1 + l * ST_FF1, b_ff1 + l * FFD, tbuf, FFD, DIMM);
    gelu_kernel<<<blocks((size_t)MROWS * FFD), 256, 0, stream>>>(tbuf, abuf, MROWS * FFD);
    gemm_bf16<true><<<dim3(DIMM / 128, MROWS / 128), 256, 0, stream>>>(
        abuf, wT_ff2 + l * ST_FF2, b_ff2 + l * DIMM, xbuf, DIMM, FFD);
  }

  // --- final LN (f32 out) ---
  ln_kernel<false><<<MROWS, 256, 0, stream>>>(xbuf, lnf_g, lnf_b, d_out);
}
